// VerifyNet_6725918785893
// MI455X (gfx1250) — matrix operand order repaired and checked by reading, not compiled, not run on hardware
//
#include <hip/hip_runtime.h>
#include <hip/hip_bf16.h>
#include <stdint.h>

#define NN   384
#define GGRP 32
#define HWSZ (256*256)
#define PIX  (NN*NN)
#define NSEG (NN+1)
#define DSEG 10

typedef __bf16 bf16_t;
typedef __attribute__((ext_vector_type(16))) __bf16 v16bf;
typedef __attribute__((ext_vector_type(8)))  __bf16 v8bf;
typedef __attribute__((ext_vector_type(8)))  float  v8f;
typedef __attribute__((ext_vector_type(8)))  unsigned int v8u;
typedef __attribute__((__vector_size__(4 * sizeof(int)))) int i32x4;

#if defined(__HIP_DEVICE_COMPILE__) && __has_builtin(__builtin_amdgcn_global_load_async_to_lds_b128)
#define USE_ASYNC_LDS 1
#else
#define USE_ASYNC_LDS 0
#endif

// ---------- bf16 helpers (native converts -> v_cvt_*bf16*) ----------
__device__ __forceinline__ bf16_t f2bf(float f) { return (bf16_t)f; }
__device__ __forceinline__ float  bf2f(bf16_t b) { return (float)b; }

// ---------- WMMA fragment loaders ----------
// A (16x32 bf16, row-major W[oc][k], ld in elements):
//   lanes 0-15 : M=lane,   K = kbase+0..7 (v0..3), kbase+16..23 (v4..7)
//   lanes 16-31: M=lane-16,K = kbase+8..15,        kbase+24..31
__device__ __forceinline__ v16bf load_pair16(const bf16_t* p0, const bf16_t* p1) {
  uint4 lo = *(const uint4*)p0;
  uint4 hi = *(const uint4*)p1;
  v8u t;
  t[0]=lo.x; t[1]=lo.y; t[2]=lo.z; t[3]=lo.w;
  t[4]=hi.x; t[5]=hi.y; t[6]=hi.z; t[7]=hi.w;
  return __builtin_bit_cast(v16bf, t);
}
__device__ __forceinline__ v16bf a_frag(const bf16_t* W, int ld, int mbase, int kbase, int lane) {
  int row  = mbase + (lane & 15);
  int koff = (lane & 16) ? 8 : 0;
  const bf16_t* p = W + (size_t)row * ld + kbase + koff;
  return load_pair16(p, p + 16);
}
// B (32x16 bf16) from pixel-major tile X[px][k]:
//   lanes 0-15 : N=lane,   K = kbase+0..7, kbase+16..23
//   lanes 16-31: N=lane-16,K = kbase+8..15, kbase+24..31
__device__ __forceinline__ v16bf b_frag(const bf16_t* X, int ld, int kbase, int lane) {
  int col = lane & 15;
  int k0  = kbase + ((lane & 16) ? 8 : 0);
  const bf16_t* p = X + (size_t)col * ld + k0;
  return load_pair16(p, p + 16);
}

// One 16-pixel GEMM tile stack: Y[16px][OC] = act(W[OC][K] @ X[16px][K] + bias)
template <bool RELU>
__device__ __forceinline__ void gemm_tile(const bf16_t* __restrict__ W, int ldw,
                                          const float*  __restrict__ bias,
                                          const bf16_t* __restrict__ X, int ldx, int K,
                                          bf16_t* __restrict__ Y, int ldy,
                                          int mtiles, int lane) {
  const int col  = lane & 15;
  const int rowb = (lane & 16) ? 8 : 0;
  for (int mt = 0; mt < mtiles; ++mt) {
    v8f acc = {};
    for (int k = 0; k < K; k += 32) {
      v16bf a = a_frag(W, ldw, mt * 16, k, lane);
      v16bf b = b_frag(X, ldx, k, lane);
      acc = __builtin_amdgcn_wmma_f32_16x16x32_bf16(false, a, false, b,
                                                    (short)0, acc, false, false);
    }
    int ocb = mt * 16 + rowb;
    float4 ba = *(const float4*)(bias + ocb);
    float4 bb = *(const float4*)(bias + ocb + 4);
    float bv[8] = {ba.x, ba.y, ba.z, ba.w, bb.x, bb.y, bb.z, bb.w};
    v8bf pk;
#pragma unroll
    for (int t = 0; t < 8; ++t) {
      float x = acc[t] + bv[t];
      if (RELU) x = fmaxf(x, 0.0f);
      pk[t] = f2bf(x);
    }
    *(v8bf*)(Y + (size_t)col * ldy + ocb) = pk;
  }
}

// ---------- small kernels ----------
__global__ void k_zero(float* p, int n) {
  int t = blockIdx.x * blockDim.x + threadIdx.x;
  if (t < n) p[t] = 0.0f;
}
__global__ void k_cvt(const float* __restrict__ s, bf16_t* __restrict__ d, int n) {
  int t = blockIdx.x * blockDim.x + threadIdx.x;
  if (t < n) d[t] = f2bf(s[t]);
}
__global__ void k_means(const float* __restrict__ src, const float* __restrict__ dst,
                        float* __restrict__ m_src, float* __restrict__ m_dst) {
  int t = blockIdx.x * blockDim.x + threadIdx.x;
  if (t >= NN * 2) return;
  int i = t >> 1, c = t & 1;
  float s = 0.f;
  for (int g = 0; g < GGRP; ++g) s += src[(i * GGRP + g) * 2 + c];
  m_src[t] = s * (1.0f / GGRP);
  s = 0.f;
  for (int g = 0; g < GGRP; ++g) s += dst[(i * GGRP + g) * 2 + c];
  m_dst[t] = s * (1.0f / GGRP);
}
__global__ void k_scatter(const float* __restrict__ src_pos, const int* __restrict__ overseg,
                          float* __restrict__ seg_sum, float* __restrict__ seg_cnt) {
  int p = blockIdx.x * blockDim.x + threadIdx.x;
  if (p >= HWSZ) return;
  int s = overseg[p];
  if (s < 0) s = NSEG - 1;
  atomicAdd(&seg_sum[s * 2 + 0], src_pos[p]);
  atomicAdd(&seg_sum[s * 2 + 1], src_pos[HWSZ + p]);
  atomicAdd(&seg_cnt[s], 1.0f);
}
__global__ void k_pos(const float* __restrict__ seg_sum, const float* __restrict__ seg_cnt,
                      float* __restrict__ pos_sp) {
  int t = blockIdx.x * blockDim.x + threadIdx.x;
  if (t >= NN) return;
  float c = fmaxf(seg_cnt[t], 1.0f);
  pos_sp[t]      = seg_sum[t * 2 + 0] / c;
  pos_sp[NN + t] = seg_sum[t * 2 + 1] / c;
}
// diff[d][i][j] = D(i,j,d)+D(j,i,d); D(i,j,d)=m_src[i,:].R[j,d,:] + T[j,d] - m_dst[i,d]
__global__ void k_diff(const float* __restrict__ m_src, const float* __restrict__ m_dst,
                       const float* __restrict__ R, const float* __restrict__ T,
                       float* __restrict__ out) {
  int p = blockIdx.x * blockDim.x + threadIdx.x;
  if (p >= PIX) return;
  int i = p / NN, j = p % NN;
#pragma unroll
  for (int d = 0; d < 2; ++d) {
    float a = m_src[i*2+0]*R[j*4+d*2+0] + m_src[i*2+1]*R[j*4+d*2+1] + T[j*2+d] - m_dst[i*2+d];
    float b = m_src[j*2+0]*R[i*4+d*2+0] + m_src[j*2+1]*R[i*4+d*2+1] + T[i*2+d] - m_dst[j*2+d];
    out[(size_t)d * PIX + p] = a + b;
  }
}

// ---------- pre MLP: [diff0,diff1,pos0(i),pos1(i)] -> 16 -> 64 (VALU) -> 512 (WMMA) ----------
__global__ void __launch_bounds__(256) k_pre(const float* __restrict__ diff,
                                             const float* __restrict__ pos_sp,
                                             const float* __restrict__ w0, const float* __restrict__ b0,
                                             const float* __restrict__ w1, const float* __restrict__ b1,
                                             const bf16_t* __restrict__ W2, const float* __restrict__ b2,
                                             bf16_t* __restrict__ U) {
  __shared__ bf16_t Xl[256 * 64];
  int tid = threadIdx.x;
  int pb  = blockIdx.x * 256;
  int p   = pb + tid;
  int i   = p / NN;
  float f[4];
  f[0] = diff[p]; f[1] = diff[PIX + p];
  f[2] = pos_sp[i]; f[3] = pos_sp[NN + i];
  float h1[16];
#pragma unroll
  for (int o = 0; o < 16; ++o) {
    float s = b0[o];
#pragma unroll
    for (int c = 0; c < 4; ++c) s += w0[o * 4 + c] * f[c];
    h1[o] = fmaxf(s, 0.0f);
  }
  for (int o = 0; o < 64; ++o) {
    float s = b1[o];
#pragma unroll
    for (int c = 0; c < 16; ++c) s += w1[o * 16 + c] * h1[c];
    Xl[tid * 64 + o] = f2bf(fmaxf(s, 0.0f));
  }
  __syncthreads();
  int wave = tid >> 5, lane = tid & 31;
  for (int g = wave; g < 16; g += 8) {
    const bf16_t* X = &Xl[g * 16 * 64];
    bf16_t* Y = U + (size_t)(pb + g * 16) * 512;
    gemm_tile<true>(W2, 64, b2, X, 64, 64, Y, 512, 32, lane);
  }
}

// ---------- max over j (axis3) -> Xg0[i][512]; max over i (axis2) -> Xg1[j][512] ----------
__global__ void __launch_bounds__(512) k_max3(const bf16_t* __restrict__ U, bf16_t* __restrict__ Xg0) {
  int i = blockIdx.x, c = threadIdx.x;
  float m = -3.0e38f;
  for (int j = 0; j < NN; ++j) m = fmaxf(m, bf2f(U[((size_t)i * NN + j) * 512 + c]));
  Xg0[(size_t)i * 512 + c] = f2bf(m);
}
__global__ void __launch_bounds__(512) k_max2(const bf16_t* __restrict__ U, bf16_t* __restrict__ Xg1) {
  int j = blockIdx.x, c = threadIdx.x;
  float m = -3.0e38f;
  for (int i = 0; i < NN; ++i) m = fmaxf(m, bf2f(U[((size_t)i * NN + j) * 512 + c]));
  Xg1[(size_t)j * 512 + c] = f2bf(m);
}

// ---------- glob MLP (512->256->256->128, relu) on 384 "pixels" for g0 and g1 ----------
__global__ void __launch_bounds__(32) k_glob(const bf16_t* __restrict__ Xg0, const bf16_t* __restrict__ Xg1,
                                             const bf16_t* __restrict__ Wg0, const float* __restrict__ bg0,
                                             const bf16_t* __restrict__ Wg1, const float* __restrict__ bg1,
                                             const bf16_t* __restrict__ Wg2, const float* __restrict__ bg2,
                                             bf16_t* __restrict__ g0, bf16_t* __restrict__ g1) {
  __shared__ bf16_t A1[16 * 256];
  __shared__ bf16_t A2[16 * 256];
  int bid = blockIdx.x;
  const bf16_t* X = (bid < 24) ? Xg0 : Xg1;
  bf16_t* G = (bid < 24) ? g0 : g1;
  int pbase = (bid % 24) * 16;
  int lane  = threadIdx.x;
  gemm_tile<true>(Wg0, 512, bg0, X + (size_t)pbase * 512, 512, 512, A1, 256, 16, lane);
  __syncthreads();
  gemm_tile<true>(Wg1, 256, bg1, A1, 256, 256, A2, 256, 16, lane);
  __syncthreads();
  gemm_tile<true>(Wg2, 256, bg2, A2, 256, 256, G + (size_t)pbase * 128, 128, 8, lane);
}

// ---------- post MLP: concat[U(512),g0(i)(128),g1(j)(128)] -> 256 -> 64 -> 16 -> 1 ----------
__global__ void __launch_bounds__(32) k_post(const bf16_t* __restrict__ U,
                                             const bf16_t* __restrict__ g0, const bf16_t* __restrict__ g1,
                                             const bf16_t* __restrict__ W1, const float* __restrict__ bb1,
                                             const bf16_t* __restrict__ W2, const float* __restrict__ bb2,
                                             const bf16_t* __restrict__ W3, const float* __restrict__ bb3,
                                             const float* __restrict__ w4, const float* __restrict__ b4,
                                             float* __restrict__ logits) {
  __shared__ bf16_t B0[16 * 768];
  __shared__ bf16_t A1[16 * 256];
  __shared__ bf16_t A2[16 * 64];
  __shared__ bf16_t A3[16 * 16];
  int lane  = threadIdx.x;
  int pbase = blockIdx.x * 16;
  __builtin_prefetch(W1 + (size_t)lane * 768, 0, 1);   // global_prefetch_b8
  for (int px = 0; px < 16; ++px) {
    int p = pbase + px;
    int i = p / NN, j = p % NN;
#if USE_ASYNC_LDS
    for (int t = lane; t < 64; t += 32)
      __builtin_amdgcn_global_load_async_to_lds_b128(
          (__attribute__((address_space(1))) i32x4*)(U + (size_t)p * 512 + t * 8),
          (__attribute__((address_space(3))) i32x4*)(&B0[px * 768 + t * 8]), 0, 0);
    if (lane < 16) {
      __builtin_amdgcn_global_load_async_to_lds_b128(
          (__attribute__((address_space(1))) i32x4*)(g0 + (size_t)i * 128 + lane * 8),
          (__attribute__((address_space(3))) i32x4*)(&B0[px * 768 + 512 + lane * 8]), 0, 0);
      __builtin_amdgcn_global_load_async_to_lds_b128(
          (__attribute__((address_space(1))) i32x4*)(g1 + (size_t)j * 128 + lane * 8),
          (__attribute__((address_space(3))) i32x4*)(&B0[px * 768 + 640 + lane * 8]), 0, 0);
    }
#else
    uint4* row = (uint4*)&B0[px * 768];
    const uint4* su = (const uint4*)(U + (size_t)p * 512);
    for (int t = lane; t < 64; t += 32) row[t] = su[t];
    const uint4* s0 = (const uint4*)(g0 + (size_t)i * 128);
    if (lane < 16) row[64 + lane] = s0[lane];
    const uint4* s1 = (const uint4*)(g1 + (size_t)j * 128);
    if (lane < 16) row[80 + lane] = s1[lane];
#endif
  }
#if USE_ASYNC_LDS
#if __has_builtin(__builtin_amdgcn_s_wait_asynccnt)
  __builtin_amdgcn_s_wait_asynccnt(0);
#else
  asm volatile("s_wait_asynccnt 0x0" ::: "memory");
#endif
#endif
  __syncthreads();
  gemm_tile<true>(W1, 768, bb1, B0, 768, 768, A1, 256, 16, lane);
  __syncthreads();
  gemm_tile<true>(W2, 256, bb2, A1, 256, 256, A2, 64, 4, lane);
  __syncthreads();
  gemm_tile<true>(W3, 64, bb3, A2, 64, 64, A3, 16, 1, lane);
  __syncthreads();
  if (lane < 16) {
    float s = b4[0];
#pragma unroll
    for (int c = 0; c < 16; ++c) s += w4[c] * bf2f(A3[lane * 16 + c]);
    logits[pbase + lane] = s;
  }
}

// ---------- spectral clustering lite: sym -> top-DSEG subspace iteration -> softmax ----------
__device__ __forceinline__ float blk_sum(float v, float* red, float* bc, int i) {
  red[i] = v;
  __syncthreads();
  if (i == 0) {
    float s = 0.f;
    for (int t = 0; t < NN; ++t) s += red[t];
    *bc = s;
  }
  __syncthreads();
  return *bc;
}
__global__ void __launch_bounds__(384) k_seg(const float* __restrict__ logits,
                                             float* __restrict__ sym,
                                             float* __restrict__ V, float* __restrict__ Y,
                                             float* __restrict__ out_seg) {
  __shared__ float red[NN];
  __shared__ float bc[1];
  int i = threadIdx.x;
  for (int j = 0; j < NN; ++j) {
    int a = i > j ? i : j, b = i > j ? j : i;
    float z = logits[a * NN + b];
    sym[i * NN + j] = 1.0f / (1.0f + __expf(-z));
  }
  for (int k = 0; k < DSEG; ++k) {
    unsigned h = (unsigned)(i * DSEG + k) * 2654435761u;
    V[i * 16 + k] = ((float)(h >> 8) * (1.0f / 16777216.0f)) - 0.5f;
  }
  __syncthreads();
  for (int it = 0; it < 16; ++it) {
    float y[DSEG];
#pragma unroll
    for (int k = 0; k < DSEG; ++k) y[k] = 0.f;
    for (int j = 0; j < NN; ++j) {
      float s = sym[i * NN + j];
#pragma unroll
      for (int k = 0; k < DSEG; ++k) y[k] += s * V[j * 16 + k];
    }
    for (int k = 0; k < DSEG; ++k) Y[i * 16 + k] = y[k];
    __syncthreads();
    for (int k = 0; k < DSEG; ++k) {
      for (int l = 0; l < k; ++l) {
        float d = blk_sum(Y[i * 16 + l] * Y[i * 16 + k], red, bc, i);
        Y[i * 16 + k] -= d * Y[i * 16 + l];
      }
      float nrm = sqrtf(blk_sum(Y[i * 16 + k] * Y[i * 16 + k], red, bc, i));
      Y[i * 16 + k] /= fmaxf(nrm, 1e-20f);
    }
    __syncthreads();
    for (int k = 0; k < DSEG; ++k) V[i * 16 + k] = Y[i * 16 + k];
    __syncthreads();
  }
  float y[DSEG];
#pragma unroll
  for (int k = 0; k < DSEG; ++k) y[k] = 0.f;
  for (int j = 0; j < NN; ++j) {
    float s = sym[i * NN + j];
#pragma unroll
    for (int k = 0; k < DSEG; ++k) y[k] += s * V[j * 16 + k];
  }
  float vp[DSEG];
  for (int k = 0; k < DSEG; ++k) {
    float lam = blk_sum(V[i * 16 + k] * y[k], red, bc, i);
    float sgn = blk_sum(V[i * 16 + k], red, bc, i);
    float scl = sqrtf(fmaxf(lam, 0.f)) * (sgn < 0.f ? -1.f : 1.f);
    float val = V[i * 16 + k] * scl;
    vp[k] = val < 0.f ? 0.f : val;
  }
  float m = vp[0];
#pragma unroll
  for (int k = 1; k < DSEG; ++k) m = fmaxf(m, vp[k]);
  float se = 0.f, e[DSEG];
#pragma unroll
  for (int k = 0; k < DSEG; ++k) { e[k] = __expf(vp[k] - m); se += e[k]; }
#pragma unroll
  for (int k = 0; k < DSEG; ++k) out_seg[i * DSEG + k] = e[k] / se;
}

// ---------- host orchestration ----------
extern "C" void kernel_launch(void* const* d_in, const int* in_sizes, int n_in,
                              void* d_out, int out_size, void* d_ws, size_t ws_size,
                              hipStream_t stream) {
  (void)in_sizes; (void)n_in; (void)out_size; (void)ws_size;
  const float* src_pg  = (const float*)d_in[0];
  const float* dst_pg  = (const float*)d_in[1];
  const float* src_pos = (const float*)d_in[2];
  const float* pred_R  = (const float*)d_in[3];
  const float* pred_T  = (const float*)d_in[4];
  const int*   overseg = (const int*)d_in[5];
  // params flattened dict-sorted: glob, post, pre; each layer (w,b)
  const float* gw0 = (const float*)d_in[6];  const float* gb0 = (const float*)d_in[7];
  const float* gw1 = (const float*)d_in[8];  const float* gb1 = (const float*)d_in[9];
  const float* gw2 = (const float*)d_in[10]; const float* gb2 = (const float*)d_in[11];
  const float* pw0 = (const float*)d_in[12]; const float* pb0 = (const float*)d_in[13];
  const float* pw1 = (const float*)d_in[14]; const float* pb1 = (const float*)d_in[15];
  const float* pw2 = (const float*)d_in[16]; const float* pb2 = (const float*)d_in[17];
  const float* pw3 = (const float*)d_in[18]; const float* pb3 = (const float*)d_in[19];
  const float* rw0 = (const float*)d_in[20]; const float* rb0 = (const float*)d_in[21];
  const float* rw1 = (const float*)d_in[22]; const float* rb1 = (const float*)d_in[23];
  const float* rw2 = (const float*)d_in[24]; const float* rb2 = (const float*)d_in[25];

  float* out        = (float*)d_out;
  float* out_diff   = out;                 // 2*PIX
  float* out_logits = out + 2 * PIX;       // PIX
  float* out_seg    = out + 3 * PIX;       // NN*DSEG

  char* base = (char*)d_ws;
  size_t off = 0;
  auto carve = [&](size_t bytes) -> char* {
    char* p = base + off;
    off += (bytes + 255) & ~(size_t)255;
    return p;
  };
  float*  m_src   = (float*)carve(NN * 2 * 4);
  float*  m_dst   = (float*)carve(NN * 2 * 4);
  float*  seg_sum = (float*)carve(NSEG * 2 * 4);
  float*  seg_cnt = (float*)carve(NSEG * 4);
  float*  pos_sp  = (float*)carve(NN * 2 * 4);
  bf16_t* Wpre2   = (bf16_t*)carve(512 * 64 * 2);
  bf16_t* Wg0b    = (bf16_t*)carve(256 * 512 * 2);
  bf16_t* Wg1b    = (bf16_t*)carve(256 * 256 * 2);
  bf16_t* Wg2b    = (bf16_t*)carve(128 * 256 * 2);
  bf16_t* Wp1b    = (bf16_t*)carve(256 * 768 * 2);
  bf16_t* Wp2b    = (bf16_t*)carve(64 * 256 * 2);
  bf16_t* Wp3b    = (bf16_t*)carve(16 * 64 * 2);
  bf16_t* Xg0     = (bf16_t*)carve((size_t)NN * 512 * 2);
  bf16_t* Xg1     = (bf16_t*)carve((size_t)NN * 512 * 2);
  bf16_t* g0b     = (bf16_t*)carve((size_t)NN * 128 * 2);
  bf16_t* g1b     = (bf16_t*)carve((size_t)NN * 128 * 2);
  float*  symb    = (float*)carve((size_t)PIX * 4);
  float*  Vb      = (float*)carve((size_t)NN * 16 * 4);
  float*  Yb      = (float*)carve((size_t)NN * 16 * 4);
  bf16_t* Ub      = (bf16_t*)carve((size_t)PIX * 512 * 2);

  k_zero<<<(NSEG * 2 + 255) / 256, 256, 0, stream>>>(seg_sum, NSEG * 2);
  k_zero<<<(NSEG + 255) / 256, 256, 0, stream>>>(seg_cnt, NSEG);
  k_means<<<(NN * 2 + 255) / 256, 256, 0, stream>>>(src_pg, dst_pg, m_src, m_dst);
  k_scatter<<<(HWSZ + 255) / 256, 256, 0, stream>>>(src_pos, overseg, seg_sum, seg_cnt);
  k_pos<<<(NN + 255) / 256, 256, 0, stream>>>(seg_sum, seg_cnt, pos_sp);
  k_cvt<<<(512 * 64 + 255) / 256, 256, 0, stream>>>(rw2, Wpre2, 512 * 64);
  k_cvt<<<(256 * 512 + 255) / 256, 256, 0, stream>>>(gw0, Wg0b, 256 * 512);
  k_cvt<<<(256 * 256 + 255) / 256, 256, 0, stream>>>(gw1, Wg1b, 256 * 256);
  k_cvt<<<(128 * 256 + 255) / 256, 256, 0, stream>>>(gw2, Wg2b, 128 * 256);
  k_cvt<<<(256 * 768 + 255) / 256, 256, 0, stream>>>(pw0, Wp1b, 256 * 768);
  k_cvt<<<(64 * 256 + 255) / 256, 256, 0, stream>>>(pw1, Wp2b, 64 * 256);
  k_cvt<<<(16 * 64 + 255) / 256, 256, 0, stream>>>(pw2, Wp3b, 16 * 64);

  k_diff<<<PIX / 256, 256, 0, stream>>>(m_src, m_dst, pred_R, pred_T, out_diff);
  k_pre<<<PIX / 256, 256, 0, stream>>>(out_diff, pos_sp, rw0, rb0, rw1, rb1, Wpre2, rb2, Ub);
  k_max3<<<NN, 512, 0, stream>>>(Ub, Xg0);
  k_max2<<<NN, 512, 0, stream>>>(Ub, Xg1);
  k_glob<<<48, 32, 0, stream>>>(Xg0, Xg1, Wg0b, gb0, Wg1b, gb1, Wg2b, gb2, g0b, g1b);
  k_post<<<PIX / 16, 32, 0, stream>>>(Ub, g0b, g1b, Wp1b, pb0, Wp2b, pb1, Wp3b, pb2,
                                      pw3, pb3, out_logits);
  k_seg<<<1, NN, 0, stream>>>(out_logits, symb, Vb, Yb, out_seg);
}
